// DCRNNNet_27573690040585
// MI455X (gfx1250) — matrix operand order repaired and checked by reading, not compiled, not run on hardware
//
#include <hip/hip_runtime.h>
#include <hip/hip_bf16.h>

// ---------------------------------------------------------------------------
// DCRNN (K=1, H0=0) forward on gfx1250 via bf16 WMMA.
// Dead code eliminated: edge_index/edge_weight unused, R-gate unused,
// only first 256 rows of each 512x256 gate weight used (fwd+bwd summed).
//
// Block = 256 threads (8 waves), 128 rows. Gate weights are staged per
// n-block into double-buffered LDS and shared by all 8 waves. B-fragment
// LDS loads use an explicit two-slot ping-pong (no rotation copies) so the
// next fragment's loads stay in flight under the current WMMAs.
// ---------------------------------------------------------------------------

typedef __attribute__((ext_vector_type(16))) __bf16 v16bf;
typedef __attribute__((ext_vector_type(8)))  __bf16 v8bf;
typedef __attribute__((ext_vector_type(8)))  float  v8f;

#define NROWS   50000
#define NPAD    50048          // multiple of 128 (rows per block)
#define CIN     256
#define CHID    256
#define COUT    64
#define KB1     8              // 256 / 32
#define NB1     16             // 256 / 16
#define KB2     8              // 256 / 32
#define OB2     4              // 64  / 16
#define HSTRIDE 272            // h-staging row stride (elems), 16B aligned
#define BL_LANE 24             // staged-B per-lane stride (elems), 48B: bank stagger
#define FRAG_EL (32 * BL_LANE) // 768 elems per staged fragment
#define BSTG_EL (16 * FRAG_EL) // 12288 elems per stage buffer (8 z + 8 h frags)
// LDS: h-staging 8*16*272*2 = 69632 B, B double-buffer 2*12288*2 = 49152 B
#define SMEM_BYTES (8 * 16 * HSTRIDE * 2 + 2 * BSTG_EL * 2)

union AFrag { v16bf v; v8bf h[2]; };

// ---------------------------------------------------------------------------
// Pack weights into WMMA B-fragment layout.
// B (32x16 bf16) packed order used here (contiguous K): lanes 0-15 hold K=0..15 (2 K per VGPR, low half =
// even K), lanes 16-31 hold K=16..31. One fragment = 32 lanes x 16 bf16.
// Gate weights: W_eff[k][j] = w[0,0][k][j] + w[1,0][k][j], k < 256.
// Linear: B(K=c, N=o) = w_lin[o][c].
// ---------------------------------------------------------------------------
__global__ void dcrnn_pack_weights(const float* __restrict__ wz,
                                   const float* __restrict__ wh,
                                   const float* __restrict__ wl,
                                   __bf16* __restrict__ pz,
                                   __bf16* __restrict__ ph,
                                   __bf16* __restrict__ pl) {
    const int f     = blockIdx.x;                // fragment id
    const int lane  = threadIdx.x;               // 0..31
    const int nloc  = lane & 15;
    const int khalf = (lane >> 4) * 16;          // 0 or 16

    if (f < 128) {                               // W_z fragments: f = kb*16 + nb
        const int kb = f >> 4, nb = f & 15;
        #pragma unroll
        for (int e = 0; e < 16; ++e) {
            const int k   = kb * 32 + khalf + e;
            const int col = nb * 16 + nloc;
            const float v = wz[k * CHID + col] + wz[512 * CHID + k * CHID + col];
            pz[(size_t)f * 512 + lane * 16 + e] = (__bf16)v;
        }
    } else if (f < 256) {                        // W_h fragments
        const int f2 = f - 128;
        const int kb = f2 >> 4, nb = f2 & 15;
        #pragma unroll
        for (int e = 0; e < 16; ++e) {
            const int k   = kb * 32 + khalf + e;
            const int col = nb * 16 + nloc;
            const float v = wh[k * CHID + col] + wh[512 * CHID + k * CHID + col];
            ph[(size_t)f2 * 512 + lane * 16 + e] = (__bf16)v;
        }
    } else {                                     // w_lin fragments: f2 = kb*4 + ob
        const int f2 = f - 256;
        const int kb = f2 >> 2, ob = f2 & 3;
        #pragma unroll
        for (int e = 0; e < 16; ++e) {
            const int c = kb * 32 + khalf + e;   // hidden channel (K)
            const int o = ob * 16 + nloc;        // out channel   (N)
            pl[(size_t)f2 * 512 + lane * 16 + e] = (__bf16)(wl[o * CHID + c]);
        }
    }
}

// ---------------------------------------------------------------------------
// Build one 8-element bf16 chunk of an A fragment from x*mask on the fly.
// ---------------------------------------------------------------------------
__device__ __forceinline__ v8bf load_a8(const float* __restrict__ xr,
                                        const float* __restrict__ mr,
                                        int off) {
    const float4 xa = *(const float4*)(xr + off);
    const float4 xb = *(const float4*)(xr + off + 4);
    const float4 ma = *(const float4*)(mr + off);
    const float4 mb = *(const float4*)(mr + off + 4);
    v8bf r;
    r[0] = (__bf16)(xa.x * ma.x); r[1] = (__bf16)(xa.y * ma.y);
    r[2] = (__bf16)(xa.z * ma.z); r[3] = (__bf16)(xa.w * ma.w);
    r[4] = (__bf16)(xb.x * mb.x); r[5] = (__bf16)(xb.y * mb.y);
    r[6] = (__bf16)(xb.z * mb.z); r[7] = (__bf16)(xb.w * mb.w);
    return r;
}

// ---------------------------------------------------------------------------
// Cooperative stage of one n-block's gate-weight fragments into LDS.
// 256 threads; thread handles fragment tid>>4 (0-7: W_z kb, 8-15: W_h kb)
// and a pair of lanes (2*(tid&15), 2*(tid&15)+1), 16 bf16 each.
// LDS lane stride = 24 elems (48 B) to stagger banks; 16 B aligned halves.
// ---------------------------------------------------------------------------
__device__ __forceinline__ void stage_b(const __bf16* __restrict__ pz,
                                        const __bf16* __restrict__ ph,
                                        __bf16* __restrict__ dst,
                                        int nb, int tid) {
    const int frag = tid >> 4;                   // 0..15
    const int sub  = tid & 15;                   // lane pair
    const __bf16* src = (frag < 8)
        ? pz + (size_t)(frag * 16 + nb) * 512
        : ph + (size_t)((frag - 8) * 16 + nb) * 512;
    const int l0 = 2 * sub, l1 = 2 * sub + 1;
    const v8bf g0 = *(const v8bf*)(src + l0 * 16);
    const v8bf g1 = *(const v8bf*)(src + l0 * 16 + 8);
    const v8bf g2 = *(const v8bf*)(src + l1 * 16);
    const v8bf g3 = *(const v8bf*)(src + l1 * 16 + 8);
    __bf16* d = dst + frag * FRAG_EL;
    *(v8bf*)(d + l0 * BL_LANE)     = g0;
    *(v8bf*)(d + l0 * BL_LANE + 8) = g1;
    *(v8bf*)(d + l1 * BL_LANE)     = g2;
    *(v8bf*)(d + l1 * BL_LANE + 8) = g3;
}

// ---------------------------------------------------------------------------
// Main fused kernel. 256 threads = 8 waves; each wave owns a 16-row strip.
// ---------------------------------------------------------------------------
__global__ __launch_bounds__(256) void dcrnn_main(
        const float* __restrict__ x,  const float* __restrict__ mask,
        const __bf16* __restrict__ pz, const __bf16* __restrict__ ph,
        const __bf16* __restrict__ pl,
        const float* __restrict__ bz, const float* __restrict__ bh,
        const float* __restrict__ blin,
        float* __restrict__ out) {

    extern __shared__ __bf16 smem[];
    __bf16* hbuf = smem;                               // 8*16*HSTRIDE
    __bf16* bst  = smem + 8 * 16 * HSTRIDE;            // 2 * BSTG_EL

    const int tid    = threadIdx.x;
    const int wave   = tid >> 5;
    const int lane   = tid & 31;
    const int laneHi = lane >> 4;                      // 0/1 half-wave
    const int col16  = lane & 15;
    const int rowTile = blockIdx.x * 128 + wave * 16;

    __bf16* myl = hbuf + wave * 16 * HSTRIDE;

    // ---- A fragments for the 16-row strip (contiguous K order, same as packed B) ----
    // lanes 0-15: row = lane,  K 0..15
    // lanes16-31: row = lane-16, K 16..31
    int rowA = rowTile + col16;
    if (rowA >= NROWS) rowA = NROWS - 1;               // clamp; padded rows unstored
    const float* xr = x    + (size_t)rowA * CIN;
    const float* mr = mask + (size_t)rowA * CIN;

    AFrag a[KB1];
    #pragma unroll
    for (int kb = 0; kb < KB1; ++kb) {
        a[kb].h[0] = load_a8(xr, mr, kb * 32 + laneHi * 16);
        a[kb].h[1] = load_a8(xr, mr, kb * 32 + 8 + laneHi * 16);
    }

    // ---- GEMM1 (Z and H_tilde) with LDS-staged, double-buffered B ----
    stage_b(pz, ph, bst, 0, tid);
    __syncthreads();

    #pragma clang loop unroll(disable)
    for (int nb = 0; nb < NB1; ++nb) {
        const __bf16* curl = bst + (nb & 1) * BSTG_EL + lane * BL_LANE;
        if (nb + 1 < NB1)
            stage_b(pz, ph, bst + ((nb + 1) & 1) * BSTG_EL, nb + 1, tid);
        if (nb + 2 < NB1) {                            // L2 hint for next stage
            __builtin_prefetch(pz + (size_t)(nb + 2) * 512, 0, 1);
            __builtin_prefetch(ph + (size_t)(nb + 2) * 512, 0, 1);
        }

        v8f az = {};
        v8f ah = {};
        // Two-slot ping-pong B fragments: slot (kb&1) consumed while slot
        // (kb&1)^1 is being loaded -- no rotation copies to coalesce.
        AFrag bzp[2], bhp[2];
        bzp[0].h[0] = *(const v8bf*)(curl);
        bzp[0].h[1] = *(const v8bf*)(curl + 8);
        bhp[0].h[0] = *(const v8bf*)(curl + 8 * FRAG_EL);
        bhp[0].h[1] = *(const v8bf*)(curl + 8 * FRAG_EL + 8);
        #pragma unroll
        for (int kb = 0; kb < KB1; ++kb) {
            const int cur = kb & 1, nxt = cur ^ 1;
            if (kb + 1 < KB1) {
                const __bf16* fz = curl + (kb + 1) * FRAG_EL;
                const __bf16* fh = curl + (8 + kb + 1) * FRAG_EL;
                bzp[nxt].h[0] = *(const v8bf*)(fz);
                bzp[nxt].h[1] = *(const v8bf*)(fz + 8);
                bhp[nxt].h[0] = *(const v8bf*)(fh);
                bhp[nxt].h[1] = *(const v8bf*)(fh + 8);
            }
            az = __builtin_amdgcn_wmma_f32_16x16x32_bf16(
                     false, a[kb].v, false, bzp[cur].v, (short)0, az, false, false);
            ah = __builtin_amdgcn_wmma_f32_16x16x32_bf16(
                     false, a[kb].v, false, bhp[cur].v, (short)0, ah, false, false);
        }

        const float bzc0 = bz[nb * 16 + col16];
        const float bhc0 = bh[nb * 16 + col16];
        // C/D layout: VGPR i -> row (i + 8*laneHi), col = lane&15
        #pragma unroll
        for (int i = 0; i < 8; ++i) {
            const float z  = 1.0f / (1.0f + __expf(-(az[i] + bzc0)));
            const float t  = tanhf(ah[i] + bhc0);
            const float Hv = (1.0f - z) * t;                   // Z*0 + (1-Z)*Ht
            const float hv = Hv > 0.0f ? Hv : (__expf(Hv) - 1.0f);  // ELU
            myl[(i + laneHi * 8) * HSTRIDE + nb * 16 + col16] = (__bf16)hv;
        }
        __syncthreads();
    }

    // ---- GEMM2: out = h @ w_lin^T + b_lin ----
    // Reload h A-fragments ONCE into the (now dead) a[] registers; reuse
    // across all 4 output column blocks. Only w_lin fragment loads remain,
    // two-slot pipelined.
    {
        const __bf16* lbase = myl + (size_t)col16 * HSTRIDE + laneHi * 16;
        #pragma unroll
        for (int kb = 0; kb < KB2; ++kb) {
            a[kb].h[0] = *(const v8bf*)(lbase + kb * 32);
            a[kb].h[1] = *(const v8bf*)(lbase + kb * 32 + 8);
        }
    }
    #pragma unroll
    for (int ob = 0; ob < OB2; ++ob) {
        v8f ac = {};
        v16bf blp[2];
        blp[0] = *(const v16bf*)(pl + ((size_t)(0 * 4 + ob) * 32 + lane) * 16);
        #pragma unroll
        for (int kb = 0; kb < KB2; ++kb) {
            const int cur = kb & 1, nxt = cur ^ 1;
            if (kb + 1 < KB2)
                blp[nxt] = *(const v16bf*)(pl + ((size_t)((kb + 1) * 4 + ob) * 32 + lane) * 16);
            ac = __builtin_amdgcn_wmma_f32_16x16x32_bf16(
                     false, a[kb].v, false, blp[cur], (short)0, ac, false, false);
        }
        const float bc = blin[ob * 16 + col16];
        #pragma unroll
        for (int i = 0; i < 8; ++i) {
            const int grow = rowTile + i + laneHi * 8;
            if (grow < NROWS)
                out[(size_t)grow * COUT + ob * 16 + col16] = ac[i] + bc;
        }
    }
}

// ---------------------------------------------------------------------------
extern "C" void kernel_launch(void* const* d_in, const int* in_sizes, int n_in,
                              void* d_out, int out_size, void* d_ws, size_t ws_size,
                              hipStream_t stream) {
    (void)in_sizes; (void)n_in; (void)out_size; (void)ws_size;

    const float* x    = (const float*)d_in[0];
    // d_in[1] edge_index, d_in[2] edge_weight: unused (K=1 diffusion)
    const float* mask = (const float*)d_in[3];
    const float* wz   = (const float*)d_in[4];
    const float* bz   = (const float*)d_in[5];
    // d_in[6] w_r, d_in[7] b_r: unused (H0 == 0 makes R a no-op)
    const float* wh   = (const float*)d_in[8];
    const float* bh   = (const float*)d_in[9];
    const float* wl   = (const float*)d_in[10];
    const float* bl   = (const float*)d_in[11];
    float* out = (float*)d_out;

    // Workspace: packed bf16 weight fragments (128 + 128 + 32 frags x 1 KB).
    char* ws = (char*)d_ws;
    __bf16* pz = (__bf16*)ws;
    __bf16* ph = pz + 128 * 512;
    __bf16* pl = ph + 128 * 512;

    dcrnn_pack_weights<<<288, 32, 0, stream>>>(wz, wh, wl, pz, ph, pl);
    dcrnn_main<<<NPAD / 128, 256, SMEM_BYTES, stream>>>(x, mask, pz, ph, pl,
                                                        bz, bh, bl, out);
}
